// VectorQuantizer_Eval_68685116998176
// MI455X (gfx1250) — matrix operand order repaired and checked by reading, not compiled, not run on hardware
//
#include <hip/hip_runtime.h>

// ---------------------------------------------------------------------------
// VQ with rotations on gfx1250 (MI455X):
//   cross[288][512] = flat[288][131072] @ emb[512][131072]^T  via
//   v_wmma_f32_16x16x32_bf16 (f32 accumulate), split-K=64 with f32 atomics.
//   Then distances d = ||x||^2 + ||e||^2 - 2*cross, per-batch argmin.
//
// Roofline: 38.7 GFLOP vs 420 MB mandatory traffic -> ~18 us memory floor at
// 23.3 TB/s; bf16 WMMA keeps compute at/below that floor (f32 WMMA would be
// ~8x slower per instruction). One wave owns a 16x64 tile; all 20 b128 loads
// of a K-step are issued before any consumer so the wave keeps deep MLP.
// ---------------------------------------------------------------------------

typedef __attribute__((ext_vector_type(16))) __bf16 v16bf;
typedef __attribute__((ext_vector_type(8)))  float  v8f;
typedef __attribute__((ext_vector_type(4)))  float  f32x4;

#define VQ_B      8
#define VQ_A      36
#define VQ_M      (VQ_B * VQ_A)       // 288 rows
#define VQ_E      512                 // codebook entries (N)
#define VQ_K      131072              // reduction dim
#define VQ_KSPLIT 64
#define VQ_KCHUNK (VQ_K / VQ_KSPLIT)  // 2048

// ---------------------------------------------------------------------------
// Zero-fill workspace (cross + norms) — harness poisons ws with 0xAA.
// ---------------------------------------------------------------------------
__global__ __launch_bounds__(256)
void vq_zero(float* __restrict__ p, int n) {
  int i = blockIdx.x * blockDim.x + threadIdx.x;
  if (i < n) p[i] = 0.0f;
}

// ---------------------------------------------------------------------------
// Row squared-norm: one 256-thread block per row, vectorized b128 loads,
// LDS tree reduce.
// ---------------------------------------------------------------------------
__global__ __launch_bounds__(256)
void vq_row_sq(const float* __restrict__ A, float* __restrict__ out) {
  __shared__ float red[256];
  const float* __restrict__ row = A + (size_t)blockIdx.x * VQ_K;
  float s = 0.0f;
  for (int i = threadIdx.x * 4; i < VQ_K; i += 256 * 4) {
    f32x4 v = *(const f32x4*)(row + i);
    s += v.x * v.x + v.y * v.y + v.z * v.z + v.w * v.w;
  }
  red[threadIdx.x] = s;
  __syncthreads();
  for (int off = 128; off > 0; off >>= 1) {
    if (threadIdx.x < (unsigned)off) red[threadIdx.x] += red[threadIdx.x + off];
    __syncthreads();
  }
  if (threadIdx.x == 0) out[blockIdx.x] = red[0];
}

// ---------------------------------------------------------------------------
// WMMA GEMM: one wave computes a 16x64 tile of cross over one K-chunk.
// grid = (M/16, E/64, KSPLIT) = (18, 8, 64), block = 32 (one wave32).
//
// A fragment as packed here (16x32, contiguous k; not the ISA table order, B matches):
//   lanes 0-15: row M=lane,    K = 0..15
//   lanes16-31: row M=lane-16, K = 16..31
// B fragment layout (16-bit B 32x16):
//   lanes 0-15: col N=lane,    K = 0..15 ; lanes16-31: col N=lane-16, K=16..31
// Both map to contiguous 128-bit global loads of the row-major f32 sources.
// ---------------------------------------------------------------------------
__global__ __launch_bounds__(32, 1)
void vq_cross_wmma(const float* __restrict__ X,
                   const float* __restrict__ Emb,
                   float* __restrict__ cross) {
  const int lane = threadIdx.x & 31;
  const int half = lane >> 4;   // 0 or 1
  const int l16  = lane & 15;

  const int  mbase = blockIdx.x * 16;
  const int  nbase = blockIdx.y * 64;
  const long kbase = (long)blockIdx.z * VQ_KCHUNK;

  const float* __restrict__ Arow = X + (size_t)(mbase + l16) * VQ_K + kbase + half * 16;
  const float* __restrict__ B0 = Emb + (size_t)(nbase +  0 + l16) * VQ_K + kbase + 16 * half;
  const float* __restrict__ B1 = Emb + (size_t)(nbase + 16 + l16) * VQ_K + kbase + 16 * half;
  const float* __restrict__ B2 = Emb + (size_t)(nbase + 32 + l16) * VQ_K + kbase + 16 * half;
  const float* __restrict__ B3 = Emb + (size_t)(nbase + 48 + l16) * VQ_K + kbase + 16 * half;

  v8f acc0 = {}, acc1 = {}, acc2 = {}, acc3 = {};

  for (int kk = 0; kk < VQ_KCHUNK; kk += 32) {
    const float* __restrict__ ap  = Arow + kk;
    const float* __restrict__ bp0 = B0 + kk;
    const float* __restrict__ bp1 = B1 + kk;
    const float* __restrict__ bp2 = B2 + kk;
    const float* __restrict__ bp3 = B3 + kk;

    // -------- issue ALL 20 b128 loads before any consumer (deep MLP) ------
    f32x4 ra0 = *(const f32x4*)(ap + 0);
    f32x4 ra1 = *(const f32x4*)(ap + 4);
    f32x4 ra2 = *(const f32x4*)(ap + 8);
    f32x4 ra3 = *(const f32x4*)(ap + 12);

    f32x4 q00 = *(const f32x4*)(bp0 + 0);
    f32x4 q01 = *(const f32x4*)(bp0 + 4);
    f32x4 q02 = *(const f32x4*)(bp0 + 8);
    f32x4 q03 = *(const f32x4*)(bp0 + 12);

    f32x4 q10 = *(const f32x4*)(bp1 + 0);
    f32x4 q11 = *(const f32x4*)(bp1 + 4);
    f32x4 q12 = *(const f32x4*)(bp1 + 8);
    f32x4 q13 = *(const f32x4*)(bp1 + 12);

    f32x4 q20 = *(const f32x4*)(bp2 + 0);
    f32x4 q21 = *(const f32x4*)(bp2 + 4);
    f32x4 q22 = *(const f32x4*)(bp2 + 8);
    f32x4 q23 = *(const f32x4*)(bp2 + 12);

    f32x4 q30 = *(const f32x4*)(bp3 + 0);
    f32x4 q31 = *(const f32x4*)(bp3 + 4);
    f32x4 q32 = *(const f32x4*)(bp3 + 8);
    f32x4 q33 = *(const f32x4*)(bp3 + 12);

    // L2 prefetch one K-step ahead of the load stream.
    __builtin_prefetch(ap  + 32, 0, 1);
    __builtin_prefetch(bp0 + 32, 0, 1);
    __builtin_prefetch(bp1 + 32, 0, 1);
    __builtin_prefetch(bp2 + 32, 0, 1);
    __builtin_prefetch(bp3 + 32, 0, 1);

    // -------- convert + WMMA, interleaved so cvts co-execute with XDL -----
    v16bf af;
    af[0]  = (__bf16)ra0.x; af[1]  = (__bf16)ra0.y; af[2]  = (__bf16)ra0.z; af[3]  = (__bf16)ra0.w;
    af[4]  = (__bf16)ra1.x; af[5]  = (__bf16)ra1.y; af[6]  = (__bf16)ra1.z; af[7]  = (__bf16)ra1.w;
    af[8]  = (__bf16)ra2.x; af[9]  = (__bf16)ra2.y; af[10] = (__bf16)ra2.z; af[11] = (__bf16)ra2.w;
    af[12] = (__bf16)ra3.x; af[13] = (__bf16)ra3.y; af[14] = (__bf16)ra3.z; af[15] = (__bf16)ra3.w;

    v16bf bf0;
    bf0[0]  = (__bf16)q00.x; bf0[1]  = (__bf16)q00.y; bf0[2]  = (__bf16)q00.z; bf0[3]  = (__bf16)q00.w;
    bf0[4]  = (__bf16)q01.x; bf0[5]  = (__bf16)q01.y; bf0[6]  = (__bf16)q01.z; bf0[7]  = (__bf16)q01.w;
    bf0[8]  = (__bf16)q02.x; bf0[9]  = (__bf16)q02.y; bf0[10] = (__bf16)q02.z; bf0[11] = (__bf16)q02.w;
    bf0[12] = (__bf16)q03.x; bf0[13] = (__bf16)q03.y; bf0[14] = (__bf16)q03.z; bf0[15] = (__bf16)q03.w;
    acc0 = __builtin_amdgcn_wmma_f32_16x16x32_bf16(false, af, false, bf0,
                                                   (short)0, acc0, false, false);

    v16bf bf1;
    bf1[0]  = (__bf16)q10.x; bf1[1]  = (__bf16)q10.y; bf1[2]  = (__bf16)q10.z; bf1[3]  = (__bf16)q10.w;
    bf1[4]  = (__bf16)q11.x; bf1[5]  = (__bf16)q11.y; bf1[6]  = (__bf16)q11.z; bf1[7]  = (__bf16)q11.w;
    bf1[8]  = (__bf16)q12.x; bf1[9]  = (__bf16)q12.y; bf1[10] = (__bf16)q12.z; bf1[11] = (__bf16)q12.w;
    bf1[12] = (__bf16)q13.x; bf1[13] = (__bf16)q13.y; bf1[14] = (__bf16)q13.z; bf1[15] = (__bf16)q13.w;
    acc1 = __builtin_amdgcn_wmma_f32_16x16x32_bf16(false, af, false, bf1,
                                                   (short)0, acc1, false, false);

    v16bf bf2;
    bf2[0]  = (__bf16)q20.x; bf2[1]  = (__bf16)q20.y; bf2[2]  = (__bf16)q20.z; bf2[3]  = (__bf16)q20.w;
    bf2[4]  = (__bf16)q21.x; bf2[5]  = (__bf16)q21.y; bf2[6]  = (__bf16)q21.z; bf2[7]  = (__bf16)q21.w;
    bf2[8]  = (__bf16)q22.x; bf2[9]  = (__bf16)q22.y; bf2[10] = (__bf16)q22.z; bf2[11] = (__bf16)q22.w;
    bf2[12] = (__bf16)q23.x; bf2[13] = (__bf16)q23.y; bf2[14] = (__bf16)q23.z; bf2[15] = (__bf16)q23.w;
    acc2 = __builtin_amdgcn_wmma_f32_16x16x32_bf16(false, af, false, bf2,
                                                   (short)0, acc2, false, false);

    v16bf bf3;
    bf3[0]  = (__bf16)q30.x; bf3[1]  = (__bf16)q30.y; bf3[2]  = (__bf16)q30.z; bf3[3]  = (__bf16)q30.w;
    bf3[4]  = (__bf16)q31.x; bf3[5]  = (__bf16)q31.y; bf3[6]  = (__bf16)q31.z; bf3[7]  = (__bf16)q31.w;
    bf3[8]  = (__bf16)q32.x; bf3[9]  = (__bf16)q32.y; bf3[10] = (__bf16)q32.z; bf3[11] = (__bf16)q32.w;
    bf3[12] = (__bf16)q33.x; bf3[13] = (__bf16)q33.y; bf3[14] = (__bf16)q33.z; bf3[15] = (__bf16)q33.w;
    acc3 = __builtin_amdgcn_wmma_f32_16x16x32_bf16(false, af, false, bf3,
                                                   (short)0, acc3, false, false);
  }

  // C/D layout: lane<16 -> rows mbase+0..7 at col nbase+l16,
  //             lane>=16 -> rows mbase+8..15. Split-K combine via f32 atomics.
  float* base = cross + (size_t)(mbase + half * 8) * VQ_E + nbase + l16;
#pragma unroll
  for (int r = 0; r < 8; ++r) {
    float* row = base + (size_t)r * VQ_E;
    atomicAdd(row + 0,  acc0[r]);
    atomicAdd(row + 16, acc1[r]);
    atomicAdd(row + 32, acc2[r]);
    atomicAdd(row + 48, acc3[r]);
  }
}

// ---------------------------------------------------------------------------
// Per-batch argmin over 36*512 distances; tie-break = lowest flat index
// (matches jnp.argmin first-occurrence semantics).
// ---------------------------------------------------------------------------
__global__ __launch_bounds__(256)
void vq_argmin(const float* __restrict__ cross,
               const float* __restrict__ xsq,
               const float* __restrict__ esq,
               float* __restrict__ out) {
  __shared__ float sv[256];
  __shared__ int   si[256];
  const int b = blockIdx.x;
  float best  = 3.402823466e38f;
  int   besti = 0x7fffffff;
  for (int i = threadIdx.x; i < VQ_A * VQ_E; i += 256) {
    const int a = i >> 9;            // i / 512
    const int e = i & (VQ_E - 1);    // i % 512
    const float d = xsq[b * VQ_A + a] + esq[e]
                  - 2.0f * cross[(size_t)(b * VQ_A + a) * VQ_E + e];
    if (d < best || (d == best && i < besti)) { best = d; besti = i; }
  }
  sv[threadIdx.x] = best;
  si[threadIdx.x] = besti;
  __syncthreads();
  for (int off = 128; off > 0; off >>= 1) {
    if (threadIdx.x < (unsigned)off) {
      float ov = sv[threadIdx.x + off];
      int   oi = si[threadIdx.x + off];
      if (ov < sv[threadIdx.x] || (ov == sv[threadIdx.x] && oi < si[threadIdx.x])) {
        sv[threadIdx.x] = ov;
        si[threadIdx.x] = oi;
      }
    }
    __syncthreads();
  }
  if (threadIdx.x == 0) {
    const int idx = si[0];
    out[b]        = (float)(idx & (VQ_E - 1)); // encoding_indices = idx % E
    out[VQ_B + b] = (float)(idx >> 9);         // best_rotations   = idx // E
  }
}

// ---------------------------------------------------------------------------
// Harness entry point.
//   d_in[0]: inputs  [8,36,32,16,16,16] f32  (= [288, 131072])
//   d_in[1]: embeddings [512, 131072] f32
//   d_out  : 16 floats (8 encoding indices, then 8 rotations)
//   d_ws   : cross [288*512] + xsq [288] + esq [512]  (f32)  ~593 KB
// ---------------------------------------------------------------------------
extern "C" void kernel_launch(void* const* d_in, const int* in_sizes, int n_in,
                              void* d_out, int out_size, void* d_ws, size_t ws_size,
                              hipStream_t stream) {
  (void)in_sizes; (void)n_in; (void)out_size; (void)ws_size;

  const float* X   = (const float*)d_in[0];
  const float* Emb = (const float*)d_in[1];

  float* cross = (float*)d_ws;
  float* xsq   = cross + (size_t)VQ_M * VQ_E;   // 288
  float* esq   = xsq + VQ_M;                    // 512
  const int ws_floats = VQ_M * VQ_E + VQ_M + VQ_E;

  vq_zero<<<(ws_floats + 255) / 256, 256, 0, stream>>>(cross, ws_floats);
  vq_row_sq<<<VQ_M, 256, 0, stream>>>(X, xsq);
  vq_row_sq<<<VQ_E, 256, 0, stream>>>(Emb, esq);
  vq_cross_wmma<<<dim3(VQ_M / 16, VQ_E / 64, VQ_KSPLIT), 32, 0, stream>>>(X, Emb, cross);
  vq_argmin<<<VQ_B, 256, 0, stream>>>(cross, xsq, esq, (float*)d_out);
}
